// CausalSelfAttention_15058155340444
// MI455X (gfx1250) — compile-verified, matrix operand order checked by reading, not run on hardware
//
#include <hip/hip_runtime.h>
#include <hip/hip_bf16.h>
#include <math.h>

// CDNA5 (gfx1250) wave32 WMMA causal self-attention, async-pipelined.
typedef _Float16 half_t;
typedef __attribute__((ext_vector_type(16))) _Float16 v16h;
typedef __attribute__((ext_vector_type(8)))  _Float16 v8h;
typedef __attribute__((ext_vector_type(4)))  _Float16 v4h;
typedef __attribute__((ext_vector_type(8)))  float    v8f;
typedef __attribute__((ext_vector_type(4)))  float    v4f;
typedef __attribute__((ext_vector_type(4)))  int      v4i;

#define T_SEQ  2048
#define N_HEAD 16
#define HEAD_D 64
#define C_DIM  1024

// V_WMMA_F32_16X16X32_F16 A/B fragment: per lane, elements 0..7 are K =
// base0..base0+7 and elements 8..15 are K = base0+16..base0+23, with
// base0 = (lane>=16 ? 8 : 0).  (ISA 7.12.2)
__device__ __forceinline__ int kbase(int lane) { return (lane & 16) ? 8 : 0; }

// Assemble a v16h fragment from two contiguous 8-half runs in LDS/global.
__device__ __forceinline__ v16h frag16(const half_t* p) {
    v8h lo = *(const v8h*)p;
    v8h hi = *(const v8h*)(p + 16);
    return __builtin_shufflevector(lo, hi, 0, 1, 2, 3, 4, 5, 6, 7,
                                   8, 9, 10, 11, 12, 13, 14, 15);
}

__device__ __forceinline__ v8f wmma_f16(v16h a, v16h b, v8f c) {
    return __builtin_amdgcn_wmma_f32_16x16x32_f16(
        /*neg_a=*/false, a, /*neg_b=*/false, b,
        /*c_mod=*/(short)0, c, /*reuse_a=*/false, /*reuse_b=*/false);
}

// ---- async global -> LDS copy (16B per lane), with synchronous fallback ----
typedef __attribute__((address_space(1))) v4i gas_v4i;  // global
typedef __attribute__((address_space(3))) v4i las_v4i;  // LDS

__device__ __forceinline__ void async_copy_b128(const half_t* g, half_t* l) {
#if __has_builtin(__builtin_amdgcn_global_load_async_to_lds_b128)
    __builtin_amdgcn_global_load_async_to_lds_b128(
        (gas_v4i*)(v4i*)(void*)g,
        (las_v4i*)(v4i*)(void*)l, 0, 0);
#else
    *(v8h*)l = *(const v8h*)g;
#endif
}
template <int N>
__device__ __forceinline__ void wait_async_lds() {
#if __has_builtin(__builtin_amdgcn_s_wait_asynccnt)
    __builtin_amdgcn_s_wait_asynccnt(N);
#else
    asm volatile("s_wait_asynccnt %0" :: "i"(N) : "memory");
#endif
}

// ---- DPP16 all-reduce across each 16-lane row (pure VALU, no LDS) ----------
template <int CTRL>
__device__ __forceinline__ float dppf(float v) {
    int s = __float_as_int(v);
    return __int_as_float(__builtin_amdgcn_update_dpp(s, s, CTRL, 0xf, 0xf, true));
}
__device__ __forceinline__ float row16_max(float x) {
    x = fmaxf(x, dppf<0xB1>(x));    // quad_perm [1,0,3,2]  (xor 1)
    x = fmaxf(x, dppf<0x4E>(x));    // quad_perm [2,3,0,1]  (xor 2)
    x = fmaxf(x, dppf<0x141>(x));   // row_half_mirror
    x = fmaxf(x, dppf<0x140>(x));   // row_mirror
    return x;
}
__device__ __forceinline__ float row16_sum(float x) {
    x += dppf<0xB1>(x);
    x += dppf<0x4E>(x);
    x += dppf<0x141>(x);
    x += dppf<0x140>(x);
    return x;
}

// ---------------------------------------------------------------------------
// One-shot conversions (streaming, tiny vs. total traffic).
// ---------------------------------------------------------------------------
__global__ __launch_bounds__(256)
void f32_to_f16_kernel(const float* __restrict__ src, half_t* __restrict__ dst)
{
    size_t i = ((size_t)blockIdx.x * 256 + threadIdx.x) * 4;
    v4f a = *(const v4f*)(src + i);
    v4h h;
    h[0] = (half_t)a[0]; h[1] = (half_t)a[1];
    h[2] = (half_t)a[2]; h[3] = (half_t)a[3];
    *(v4h*)(dst + i) = h;
}

// Wt[n*K + k] = (f16) W[k*N + n];  K == C_DIM (compile-time shifts).
__global__ __launch_bounds__(256)
void wt_transpose_kernel(const float* __restrict__ W, half_t* __restrict__ Wt, int N)
{
    size_t i = (size_t)blockIdx.x * 256 + threadIdx.x;   // over N*K
    int n = (int)(i >> 10);
    int k = (int)(i & (C_DIM - 1));
    Wt[i] = (half_t)W[(size_t)k * N + n];
}

// ---------------------------------------------------------------------------
// Tiled GEMM: out = A[M,K](f16) @ W[K,N] + bias[N], W pre-transposed f16 Wt[N][K].
// MODE 0 : QKV epilogue -> q,k f16 [B,NH,T,HD]; v f16 [B,NH,HD,T] (transposed);
//          k/v also to d_out (f32, [B,NH,T,HD]).
// MODE 1 : plain row-major f32 store (output projection).
// Block = 128 threads (4 waves), 64x64 tile, Kstep 64, double-buffered async
// staging (8 async b128 per thread per tile). 8 WMMA per wave per K-step.
// ---------------------------------------------------------------------------
__device__ __forceinline__ void gemm_stage(const half_t* __restrict__ A,
                                           const half_t* __restrict__ Wt,
                                           half_t* AsB, half_t* BsB,
                                           int m0, int n0, int kk, int K, int tid)
{
#pragma unroll
    for (int u = 0; u < 4; ++u) {
        int off = (tid + u * 128) * 8;       // 4096 halves per tile
        int r = off >> 6, c = off & 63;
        async_copy_b128(A  + (size_t)(m0 + r) * K + kk + c, AsB + off);
        async_copy_b128(Wt + (size_t)(n0 + r) * K + kk + c, BsB + off);
    }
}

template <int MODE>
__global__ __launch_bounds__(128)
void gemm_wmma_kernel(const half_t* __restrict__ A, const half_t* __restrict__ Wt,
                      const float* __restrict__ bias,
                      half_t* __restrict__ qh, half_t* __restrict__ kh,
                      half_t* __restrict__ vh,
                      float* __restrict__ out_y, float* __restrict__ out_k,
                      float* __restrict__ out_v,
                      int N, int K)
{
    __shared__ __align__(16) half_t As[2 * 64 * 64];   // [buf][m][k]
    __shared__ __align__(16) half_t Bst[2 * 64 * 64];  // [buf][n][k]

    const int tid  = threadIdx.x;
    const int lane = tid & 31;
    const int wave = tid >> 5;       // 0..3
    const int wm   = wave >> 1;      // 0..1
    const int wn   = wave & 1;       // 0..1
    const int m0   = blockIdx.x * 64;
    const int n0   = blockIdx.y * 64;
    const int kb   = kbase(lane);

    v8f acc[2][2];
#pragma unroll
    for (int a = 0; a < 2; ++a)
#pragma unroll
        for (int b = 0; b < 2; ++b) acc[a][b] = (v8f)(0.0f);

    const int NIT = K >> 6;
    gemm_stage(A, Wt, As, Bst, m0, n0, 0, K, tid);     // prologue: tile 0

    for (int it = 0; it < NIT; ++it) {
        half_t* AsB = As  + (it & 1) * 4096;
        half_t* BsB = Bst + (it & 1) * 4096;
        __syncthreads();                               // prev compute done
        if (it + 1 < NIT) {
            gemm_stage(A, Wt, As + ((it + 1) & 1) * 4096,
                       Bst + ((it + 1) & 1) * 4096, m0, n0, (it + 1) << 6, K, tid);
            wait_async_lds<8>();                       // tile it complete
        } else {
            wait_async_lds<0>();
        }
        __syncthreads();                               // publish tile it

#pragma unroll
        for (int ks = 0; ks < 2; ++ks) {
            v16h afr[2], bfr[2];
#pragma unroll
            for (int mt = 0; mt < 2; ++mt) {
                int row = wm * 32 + mt * 16 + (lane & 15);
                afr[mt] = frag16(AsB + row * 64 + ks * 32 + kb);
            }
#pragma unroll
            for (int nt = 0; nt < 2; ++nt) {
                int col = wn * 32 + nt * 16 + (lane & 15);
                bfr[nt] = frag16(BsB + col * 64 + ks * 32 + kb);
            }
#pragma unroll
            for (int mt = 0; mt < 2; ++mt)
#pragma unroll
                for (int nt = 0; nt < 2; ++nt)
                    acc[mt][nt] = wmma_f16(afr[mt], bfr[nt], acc[mt][nt]);
        }
    }

    // Epilogue. C/D layout: element j of lane L -> M = j + (L>=16?8:0), N = L&15.
#pragma unroll
    for (int mt = 0; mt < 2; ++mt)
#pragma unroll
        for (int nt = 0; nt < 2; ++nt)
#pragma unroll
            for (int j = 0; j < 8; ++j) {
                int m = m0 + wm * 32 + mt * 16 + j + ((lane & 16) ? 8 : 0);
                int n = n0 + wn * 32 + nt * 16 + (lane & 15);
                float val = acc[mt][nt][j] + bias[n];
                if constexpr (MODE == 1) {
                    out_y[(size_t)m * N + n] = val;
                } else {
                    int sec = n >> 10;          // 0=q 1=k 2=v
                    int c   = n & (C_DIM - 1);
                    int h   = c >> 6;
                    int hd  = c & (HEAD_D - 1);
                    int bb  = m >> 11;          // m / T_SEQ
                    int t   = m & (T_SEQ - 1);
                    size_t di = (((size_t)bb * N_HEAD + h) * T_SEQ + t) * HEAD_D + hd;
                    if (sec == 0) {
                        qh[di] = (half_t)val;
                    } else if (sec == 1) {
                        out_k[di] = val; kh[di] = (half_t)val;
                    } else {
                        out_v[di] = val;
                        // f16 V stored transposed [B,NH,HD,T] for async staging
                        size_t dv = (((size_t)bb * N_HEAD + h) * HEAD_D + hd) * T_SEQ + t;
                        vh[dv] = (half_t)val;
                    }
                }
            }
}

// ---------------------------------------------------------------------------
// Flash-style causal attention. Block = 128 threads = 4 waves, 64 queries per
// block (16 per wave), one (b, head) per block. Key blocks of 32, double-
// buffered: K tile [key][hd] and pre-transposed V tile [hd][key] are both
// pure async b128 copies (4 async instructions per tile, zero staging VALU).
// Softmax row stats via DPP16. Output f16 [B,T,C].
// ---------------------------------------------------------------------------
__device__ __forceinline__ void attn_stage(const half_t* __restrict__ kp,
                                           const half_t* __restrict__ vp,
                                           half_t* KsB, half_t* VstB,
                                           int key0, int tid)
{
#pragma unroll
    for (int u = 0; u < 2; ++u) {            // K tile: 2048 halves [key][hd]
        int off = (tid + u * 128) * 8;
        int r = off >> 6, c = off & 63;
        async_copy_b128(kp + (size_t)(key0 + r) * HEAD_D + c, KsB + off);
    }
#pragma unroll
    for (int u = 0; u < 2; ++u) {            // V tile: 2048 halves [hd][key]
        int off = (tid + u * 128) * 8;
        int hd = off >> 5, c = off & 31;
        async_copy_b128(vp + (size_t)hd * T_SEQ + key0 + c, VstB + off);
    }
}

__global__ __launch_bounds__(128)
void attn_wmma_kernel(const half_t* __restrict__ qh,
                      const half_t* __restrict__ kh,
                      const half_t* __restrict__ vh,
                      half_t* __restrict__ attn_h)
{
    __shared__ __align__(16) half_t Ks[2 * 32 * 64];   // [buf][key][hd]
    __shared__ __align__(16) half_t Vst[2 * 64 * 32];  // [buf][hd][key]
    __shared__ __align__(16) half_t Ps[4 * 16 * 32];   // per-wave [row][key]

    const int tid  = threadIdx.x;
    const int lane = tid & 31;
    const int wave = tid >> 5;
    const int kb   = kbase(lane);

    const int qblk = blockIdx.x & 31;            // T/64 = 32 query blocks
    const int h    = (blockIdx.x >> 5) & 15;
    const int b    = blockIdx.x >> 9;
    const int q0   = qblk * 64;
    const int qw   = q0 + wave * 16;             // this wave's first query row

    const size_t bh = ((size_t)b * N_HEAD + h) * T_SEQ;
    const half_t* qp = qh + bh * HEAD_D;
    const half_t* kp = kh + bh * HEAD_D;
    const half_t* vp = vh + bh * HEAD_D;         // [hd][t] layout, same flat base

    // Q fragments (two K-steps over HD=64): pure vector loads from global.
    v16h aq[2];
    {
        int row = qw + (lane & 15);
#pragma unroll
        for (int ks = 0; ks < 2; ++ks)
            aq[ks] = frag16(qp + (size_t)row * HEAD_D + ks * 32 + kb);
    }

    const float scale = 0.125f;                  // 1/sqrt(HD), applied to S
    float m_i[8], l_i[8];
    v8f o[4];
#pragma unroll
    for (int j = 0; j < 8; ++j) { m_i[j] = -INFINITY; l_i[j] = 0.0f; }
#pragma unroll
    for (int nt = 0; nt < 4; ++nt) o[nt] = (v8f)(0.0f);

    const int nkb = qblk * 2 + 2;                // key blocks covering keys <= q0+63
    attn_stage(kp, vp, Ks, Vst, 0, tid);         // prologue: tile 0

    for (int jb = 0; jb < nkb; ++jb) {
        const int key0 = jb * 32;
        half_t* KsB  = Ks  + (jb & 1) * 2048;
        half_t* VstB = Vst + (jb & 1) * 2048;
        __syncthreads();                         // prev compute done
        if (jb + 1 < nkb) {
            attn_stage(kp, vp, Ks + ((jb + 1) & 1) * 2048,
                       Vst + ((jb + 1) & 1) * 2048, key0 + 32, tid);
            wait_async_lds<4>();                 // tile jb complete
        } else {
            wait_async_lds<0>();
        }
        __syncthreads();                         // publish tile jb

        if (key0 > qw + 15) continue;            // fully masked for this wave (uniform)

        // S = Q K^T : 2 key tiles x 2 K-steps.
        v8f s[2];
#pragma unroll
        for (int kt = 0; kt < 2; ++kt) {
            s[kt] = (v8f)(0.0f);
            int col = kt * 16 + (lane & 15);     // key index within block
#pragma unroll
            for (int ks = 0; ks < 2; ++ks) {
                v16h bk = frag16(KsB + col * 64 + ks * 32 + kb);
                s[kt] = wmma_f16(aq[ks], bk, s[kt]);
            }
        }

        // Scale + causal mask (finite sentinel so exp() stays clean).
#pragma unroll
        for (int kt = 0; kt < 2; ++kt)
#pragma unroll
            for (int j = 0; j < 8; ++j) {
                int qg = qw + j + ((lane & 16) ? 8 : 0);
                int kg = key0 + kt * 16 + (lane & 15);
                s[kt][j] = (kg > qg) ? -1.0e30f : s[kt][j] * scale;
            }

        // Online softmax: row stats across 16-lane halves via DPP (VALU only).
        float alpha[8];
#pragma unroll
        for (int j = 0; j < 8; ++j) {
            float rm = row16_max(fmaxf(s[0][j], s[1][j]));
            float mnew = fmaxf(m_i[j], rm);
            alpha[j] = expf(m_i[j] - mnew);
            float p0 = expf(s[0][j] - mnew);
            float p1 = expf(s[1][j] - mnew);
            s[0][j] = p0; s[1][j] = p1;
            float rs = row16_sum(p0 + p1);
            l_i[j] = l_i[j] * alpha[j] + rs;
            m_i[j] = mnew;
        }
#pragma unroll
        for (int nt = 0; nt < 4; ++nt)
#pragma unroll
            for (int j = 0; j < 8; ++j) o[nt][j] *= alpha[j];

        // P (C-layout) -> per-wave LDS [row][key] -> A-layout fragment.
        // Same-wave DS ops are processed in order; wave_barrier pins compiler.
        half_t* myP = &Ps[wave * 512];
#pragma unroll
        for (int kt = 0; kt < 2; ++kt)
#pragma unroll
            for (int j = 0; j < 8; ++j) {
                int row = j + ((lane & 16) ? 8 : 0);
                int key = kt * 16 + (lane & 15);
                myP[row * 32 + key] = (half_t)s[kt][j];
            }
        __builtin_amdgcn_wave_barrier();

        v16h ap = frag16(myP + (lane & 15) * 32 + kb);

        // O += P (16x32) @ V (32x64): 4 hd tiles, K=32 keys = one WMMA each.
#pragma unroll
        for (int nt = 0; nt < 4; ++nt) {
            v16h bv = frag16(VstB + (nt * 16 + (lane & 15)) * 32 + kb);
            o[nt] = wmma_f16(ap, bv, o[nt]);
        }
    }

    // Finalize: O /= l, write f16 [B, T, NH*HD] for the output projection.
    float inv[8];
#pragma unroll
    for (int j = 0; j < 8; ++j) inv[j] = 1.0f / l_i[j];
#pragma unroll
    for (int nt = 0; nt < 4; ++nt)
#pragma unroll
        for (int j = 0; j < 8; ++j) {
            int qg = qw + j + ((lane & 16) ? 8 : 0);
            int c  = h * HEAD_D + nt * 16 + (lane & 15);
            attn_h[((size_t)b * T_SEQ + qg) * C_DIM + c] = (half_t)(o[nt][j] * inv[j]);
        }
}

// ---------------------------------------------------------------------------
extern "C" void kernel_launch(void* const* d_in, const int* in_sizes, int n_in,
                              void* d_out, int out_size, void* d_ws, size_t ws_size,
                              hipStream_t stream) {
    (void)in_sizes; (void)n_in; (void)out_size; (void)ws_size;
    const float* x     = (const float*)d_in[0];   // [B,T,C]
    const float* Wqkv  = (const float*)d_in[1];   // [C,3C]
    const float* bqkv  = (const float*)d_in[2];   // [3C]
    const float* Wout  = (const float*)d_in[3];   // [C,C]
    const float* bout  = (const float*)d_in[4];   // [C]

    const size_t YN = (size_t)4 * T_SEQ * C_DIM;  // 8388608
    float* y    = (float*)d_out;                  // [B,T,C]
    float* kout = y + YN;                         // [B,NH,T,HD] f32 (output)
    float* vout = y + 2 * YN;                     // [B,NH,T,HD] f32 (output)

    half_t* qh     = (half_t*)d_ws;               // [B,NH,T,HD] f16
    half_t* khalf  = qh + YN;                     // [B,NH,T,HD] f16
    half_t* vhalf  = qh + 2 * YN;                 // [B,NH,HD,T] f16 (transposed)
    half_t* ah     = qh + 3 * YN;                 // [B,T,C] f16; doubles as xh
    half_t* wqkvt  = qh + 4 * YN;                 // [3C][C] f16 (transposed)
    half_t* woutt  = wqkvt + (size_t)3 * C_DIM * C_DIM;  // [C][C] f16 (transposed)
    half_t* xh     = ah;   // x(f16) lives in ah: consumed by gemm0 before attn writes ah

    dim3 blk(128);
    // One-shot conversions (x -> f16; weights -> f16 transposed).
    f32_to_f16_kernel<<<dim3((int)(YN / 1024)), dim3(256), 0, stream>>>(x, xh);
    wt_transpose_kernel<<<dim3((3 * C_DIM * C_DIM) / 256), dim3(256), 0, stream>>>(
        Wqkv, wqkvt, 3 * C_DIM);
    wt_transpose_kernel<<<dim3((C_DIM * C_DIM) / 256), dim3(256), 0, stream>>>(
        Wout, woutt, C_DIM);
    // QKV projection: M=8192, N=3072, K=1024
    gemm_wmma_kernel<0><<<dim3(128, 48), blk, 0, stream>>>(
        xh, wqkvt, bqkv, qh, khalf, vhalf, nullptr, kout, vout, 3 * C_DIM, C_DIM);
    // Attention: B*NH*(T/64) = 2048 blocks
    attn_wmma_kernel<<<dim3(2048), blk, 0, stream>>>(qh, khalf, vhalf, ah);
    // Output projection: M=8192, N=1024, K=1024
    gemm_wmma_kernel<1><<<dim3(128, 16), blk, 0, stream>>>(
        ah, woutt, bout, nullptr, nullptr, nullptr, y, nullptr, nullptr, C_DIM, C_DIM);
}
